// LocalScopeSelfAttention_22565758173962
// MI455X (gfx1250) — matrix operand order repaired and checked by reading, not compiled, not run on hardware
//
#include <hip/hip_runtime.h>

// ---------------- types for WMMA fragments ----------------
typedef __bf16  bf16x8  __attribute__((ext_vector_type(8)));
typedef __bf16  bf16x16 __attribute__((ext_vector_type(16)));
typedef float   f32x8   __attribute__((ext_vector_type(8)));

static __device__ __forceinline__ bf16x16 mk_frag(bf16x8 lo, bf16x8 hi) {
    return __builtin_shufflevector(lo, hi, 0,1,2,3,4,5,6,7,8,9,10,11,12,13,14,15);
}

// ---------------- problem constants ----------------
#define DIM      256
#define NHEADS   8
#define HDIM     32
#define HH       32
#define WW       32
#define NTOK     1024            // HH*WW per (b,t) slice
#define BT       16              // B*T
#define MROWS    (BT * NTOK)     // 16384 total token rows
#define MTILES   (MROWS / 16)    // 1024

// workspace layout (bytes)
#define WT_ELEMS   (4 * DIM * DIM)                     // 4 transposed bf16 weight mats
#define QKV_ELEMS  (3 * BT * NHEADS * NTOK * HDIM)     // bf16 q,k,v head-major
#define AOUT_ELEMS ((size_t)MROWS * DIM)               // bf16 attention output
#define OFF_QKV    ((size_t)WT_ELEMS * 2)
#define OFF_AOUT   (OFF_QKV + (size_t)QKV_ELEMS * 2)

// =====================================================================
// Kernel 0: weights fp32 [k][n]  ->  bf16 transposed [mat][n][k]
// =====================================================================
__global__ void k_prep(const float* __restrict__ wq, const float* __restrict__ wk,
                       const float* __restrict__ wv, const float* __restrict__ wo,
                       __bf16* __restrict__ wt) {
    int i = blockIdx.x * blockDim.x + threadIdx.x;        // 0 .. 4*65536-1
    int mat = i >> 16;
    int e   = i & 0xFFFF;                                 // e = k*256 + n
    int k = e >> 8, n = e & 255;
    const float* w = (mat == 0) ? wq : (mat == 1) ? wk : (mat == 2) ? wv : wo;
    wt[((size_t)mat << 16) + n * DIM + k] = (__bf16)w[e];
}

// =====================================================================
// Kernel 1: fused LayerNorm + QKV projection (bf16 WMMA, fp32 accum)
//   grid: 1024 blocks (one 16-row M-tile each), block: 256 threads (8 waves)
// =====================================================================
__global__ void k_ln_qkv(const float* __restrict__ x,
                         const float* __restrict__ ln_g, const float* __restrict__ ln_b,
                         const __bf16* __restrict__ wt,          // [3][n][k] bf16
                         const float* __restrict__ bq, const float* __restrict__ bk,
                         const float* __restrict__ bv,
                         __bf16* __restrict__ qkv) {             // [3][bt][h][n][32] bf16
    const int mbase = blockIdx.x * 16;
    const int t   = threadIdx.x;
    const int r   = t >> 4;          // row in tile, 0..15
    const int seg = t & 15;          // 16-col segment, 0..15

    __shared__ __bf16 xn_s[16][DIM + 8];     // +8 bf16 pad -> 528B row stride, b128 conflict-free
    __shared__ float  red_s[16][16];
    __shared__ float  red_q[16][16];

    // ---- LayerNorm: each thread owns 16 contiguous elements of one row ----
    const float4* xr4 = reinterpret_cast<const float4*>(
        x + (size_t)(mbase + r) * DIM + seg * 16);
    float vbuf[16];
    float s = 0.f, ss = 0.f;
#pragma unroll
    for (int i = 0; i < 4; ++i) {
        float4 f = xr4[i];
        vbuf[4*i+0] = f.x; vbuf[4*i+1] = f.y; vbuf[4*i+2] = f.z; vbuf[4*i+3] = f.w;
        s  += f.x + f.y + f.z + f.w;
        ss += f.x*f.x + f.y*f.y + f.z*f.z + f.w*f.w;
    }
    red_s[r][seg] = s;
    red_q[r][seg] = ss;
    __syncthreads();
    float sum = 0.f, sq = 0.f;
#pragma unroll
    for (int i = 0; i < 16; ++i) { sum += red_s[r][i]; sq += red_q[r][i]; }
    const float mu   = sum * (1.f / DIM);
    const float var  = sq * (1.f / DIM) - mu * mu;
    const float rstd = rsqrtf(var + 1e-5f);
#pragma unroll
    for (int i = 0; i < 16; ++i) {
        int c = seg * 16 + i;
        float xn = (vbuf[i] - mu) * rstd * ln_g[c] + ln_b[c];
        xn_s[r][c] = (__bf16)xn;
    }
    __syncthreads();

    // ---- WMMA GEMM: wave w -> output cols [32w, 32w+32) of q,k,v ----
    const int wave = t >> 5, lane = t & 31;
    const int ln16 = lane & 15, hi = lane >> 4;
    const int colBase = wave * 32;

    f32x8 acc[3][2] = {};
#pragma unroll
    for (int ks = 0; ks < 8; ++ks) {
        const int kb = ks * 32;
        // A fragment: 16x32 bf16 tile of xn (ISA layout: halves at kb+8*hi, kb+16+8*hi)
        const __bf16* ap = &xn_s[ln16][kb + 8 * hi];
        bf16x16 afrag = mk_frag(*(const bf16x8*)ap, *(const bf16x8*)(ap + 16));
#pragma unroll
        for (int m = 0; m < 3; ++m) {
#pragma unroll
            for (int ct = 0; ct < 2; ++ct) {
                const int n = colBase + ct * 16 + ln16;
                // B fragment: 32x16 bf16; N-major weights -> halves at kb+8*hi, kb+16+8*hi
                const __bf16* bp = wt + (((size_t)m * DIM + n) * DIM) + kb + 8 * hi;
                bf16x16 bfrag = mk_frag(*(const bf16x8*)bp, *(const bf16x8*)(bp + 16));
                acc[m][ct] = __builtin_amdgcn_wmma_f32_16x16x32_bf16(
                    false, afrag, false, bfrag, (short)0, acc[m][ct], false, false);
            }
        }
    }

    // ---- bias + store bf16, head-major [m][bt][head][n][32] ----
    const float* biases[3] = {bq, bk, bv};
    const int bt_idx   = mbase >> 10;       // slice
    const int nInSlice = mbase & 1023;      // base row within slice
#pragma unroll
    for (int m = 0; m < 3; ++m) {
#pragma unroll
        for (int ct = 0; ct < 2; ++ct) {
            const int col  = colBase + ct * 16 + ln16;
            const float bb = biases[m][col];
            const int head = col >> 5, d = col & 31;
#pragma unroll
            for (int vr = 0; vr < 8; ++vr) {
                const int row = nInSlice + vr + 8 * hi;
                size_t o = ((((size_t)m * BT + bt_idx) * NHEADS + head) * NTOK + row) * HDIM + d;
                qkv[o] = (__bf16)(acc[m][ct][vr] + bb);
            }
        }
    }
}

// =====================================================================
// Kernel 2: 3x3 local-window attention. One wave32 per (token, head);
//   lanes = head dim. Neighbor indices computed inline (clip to edges).
// =====================================================================
__global__ void k_attn(const __bf16* __restrict__ qkv,
                       __bf16* __restrict__ aout) {            // [16384][256] bf16
    const int gwave = (blockIdx.x * blockDim.x + threadIdx.x) >> 5;
    const int lane  = threadIdx.x & 31;
    const int head  = gwave & (NHEADS - 1);
    const int tok   = gwave >> 3;            // 0..16383
    const int bt    = tok >> 10, n = tok & 1023;
    const int hh = n >> 5, ww = n & 31;

    const size_t hbase = (((size_t)bt) * NHEADS + head) * NTOK * HDIM;
    const __bf16* qp = qkv + hbase + (size_t)n * HDIM;
    const __bf16* kp = qkv + (size_t)QKV_ELEMS / 3 + hbase;     // k matrix
    const __bf16* vp = qkv + 2 * ((size_t)QKV_ELEMS / 3) + hbase;

    const float qd = (float)qp[lane];
    float sc[9];
    int   nb[9];
#pragma unroll
    for (int j = 0; j < 9; ++j) {
        const int dh = j / 3 - 1, dw = j % 3 - 1;
        const int nh = min(max(hh + dh, 0), HH - 1);
        const int nw = min(max(ww + dw, 0), WW - 1);
        const int nn = nh * WW + nw;
        nb[j] = nn;
        float p = qd * (float)kp[(size_t)nn * HDIM + lane];
#pragma unroll
        for (int off = 16; off; off >>= 1) p += __shfl_xor(p, off, 32);
        sc[j] = p * 0.17677669529663687f;    // 1/sqrt(32)
    }
    float mx = sc[0];
#pragma unroll
    for (int j = 1; j < 9; ++j) mx = fmaxf(mx, sc[j]);
    float e[9], den = 0.f;
#pragma unroll
    for (int j = 0; j < 9; ++j) { e[j] = __expf(sc[j] - mx); den += e[j]; }
    const float inv = 1.f / den;
    float o = 0.f;
#pragma unroll
    for (int j = 0; j < 9; ++j)
        o += e[j] * inv * (float)vp[(size_t)nb[j] * HDIM + lane];

    aout[(size_t)tok * DIM + head * HDIM + lane] = (__bf16)o;
}

// =====================================================================
// Kernel 3: output projection (WMMA) + bias + residual, fp32 store
// =====================================================================
__global__ void k_proj_out(const __bf16* __restrict__ aout,    // [16384][256] bf16
                           const __bf16* __restrict__ wo_t,    // [n][k] bf16
                           const float* __restrict__ bo,
                           const float* __restrict__ x,
                           float* __restrict__ out) {
    const int mbase = blockIdx.x * 16;
    const int t = threadIdx.x;
    const int wave = t >> 5, lane = t & 31;
    const int ln16 = lane & 15, hi = lane >> 4;
    const int colBase = wave * 32;

    f32x8 acc[2] = {};
#pragma unroll
    for (int ks = 0; ks < 8; ++ks) {
        const int kb = ks * 32;
        const __bf16* ap = aout + (size_t)(mbase + ln16) * DIM + kb + 8 * hi;
        bf16x16 afrag = mk_frag(*(const bf16x8*)ap, *(const bf16x8*)(ap + 16));
#pragma unroll
        for (int ct = 0; ct < 2; ++ct) {
            const int n = colBase + ct * 16 + ln16;
            const __bf16* bp = wo_t + (size_t)n * DIM + kb + 8 * hi;
            bf16x16 bfrag = mk_frag(*(const bf16x8*)bp, *(const bf16x8*)(bp + 16));
            acc[ct] = __builtin_amdgcn_wmma_f32_16x16x32_bf16(
                false, afrag, false, bfrag, (short)0, acc[ct], false, false);
        }
    }
#pragma unroll
    for (int ct = 0; ct < 2; ++ct) {
        const int col = colBase + ct * 16 + ln16;
        const float bb = bo[col];
#pragma unroll
        for (int vr = 0; vr < 8; ++vr) {
            const int row = mbase + vr + 8 * hi;
            const size_t o = (size_t)row * DIM + col;
            out[o] = acc[ct][vr] + bb + x[o];
        }
    }
}

// =====================================================================
extern "C" void kernel_launch(void* const* d_in, const int* in_sizes, int n_in,
                              void* d_out, int out_size, void* d_ws, size_t ws_size,
                              hipStream_t stream) {
    const float* x    = (const float*)d_in[0];
    const float* ln_g = (const float*)d_in[1];
    const float* ln_b = (const float*)d_in[2];
    const float* wq   = (const float*)d_in[3];
    const float* bq   = (const float*)d_in[4];
    const float* wk   = (const float*)d_in[5];
    const float* bk   = (const float*)d_in[6];
    const float* wv   = (const float*)d_in[7];
    const float* bv   = (const float*)d_in[8];
    const float* wo   = (const float*)d_in[9];
    const float* bo   = (const float*)d_in[10];
    float* out = (float*)d_out;

    char* ws = (char*)d_ws;
    __bf16* wt   = (__bf16*)ws;                       // 4 transposed bf16 weights
    __bf16* wo_t = wt + 3 * DIM * DIM;
    __bf16* qkv  = (__bf16*)(ws + OFF_QKV);
    __bf16* aout = (__bf16*)(ws + OFF_AOUT);

    k_prep<<<(4 * DIM * DIM) / 256, 256, 0, stream>>>(wq, wk, wv, wo, wt);
    k_ln_qkv<<<MTILES, 256, 0, stream>>>(x, ln_g, ln_b, wt, bq, bk, bv, qkv);
    k_attn<<<(MROWS * NHEADS) / 8, 256, 0, stream>>>(qkv, aout);
    k_proj_out<<<MTILES, 256, 0, stream>>>(aout, wo_t, bo, x, out);
}
